// Attention_39943195853259
// MI455X (gfx1250) — compile-verified, matrix operand order checked by reading, not run on hardware
//
#include <hip/hip_runtime.h>
#include <hip/hip_bf16.h>

typedef __attribute__((ext_vector_type(8)))  _Float16 v8h;
typedef __attribute__((ext_vector_type(16))) _Float16 v16h;
typedef __attribute__((ext_vector_type(8)))  float    v8f;

#define C_IN 512
#define SEQ  1024
#define NH   8
#define DHEAD 64

// ---------------------------------------------------------------------------
// helpers
// ---------------------------------------------------------------------------

// Load a 16x32 f16 fragment (A or B operand) from an LDS tile stored
// K-contiguous: tile[row][k], row stride = STRIDE halfs.
// Layout per CDNA5 ISA 7.12.2 (16-bit A 16x32): lane holds row (lane&15),
// K base = (lane>=16)?8:0, chunks [kb,kb+8) in v0-3 and [kb+16,kb+24) in v4-7.
template<int STRIDE>
__device__ __forceinline__ v16h ldfrag(const _Float16* tile, int lane, int kcol) {
    const _Float16* p = tile + (lane & 15) * STRIDE + ((lane >> 4) << 3) + kcol;
    union { v16h v; v8h h[2]; } u;
    u.h[0] = *(const v8h*)(p);
    u.h[1] = *(const v8h*)(p + 16);
    return u.v;
}

__device__ __forceinline__ v8f wmma_f16(v16h a, v16h b, v8f c) {
    return __builtin_amdgcn_wmma_f32_16x16x32_f16(
        /*neg_a=*/false, a, /*neg_b=*/false, b,
        /*c_mod=*/(short)0, c, /*reuse_a=*/false, /*reuse_b=*/false);
}

// Async global -> LDS copy, GVS mode: SGPR64 base + per-lane 32-bit byte
// offset (CDNA5 GLOBAL_LOAD_ASYNC_TO_LDS_B128, ASYNCcnt).
__device__ __forceinline__ void async_b128s(void* lds_dst, const void* base,
                                            int byte_off) {
    asm volatile("global_load_async_to_lds_b128 %0, %1, %2"
                 :: "v"((unsigned int)(uintptr_t)lds_dst), "v"(byte_off),
                    "s"(base)
                 : "memory");
}
__device__ __forceinline__ void wait_async0() {
    asm volatile("s_wait_asynccnt 0" ::: "memory");
}
__device__ __forceinline__ void wait_async4() {
    asm volatile("s_wait_asynccnt 4" ::: "memory");
}
__device__ __forceinline__ void wait_async8() {
    asm volatile("s_wait_asynccnt 8" ::: "memory");
}

// ---------------------------------------------------------------------------
// Kernel 0: tile transpose + f32->f16 convert.  in[R=512][Cdim] f32 ->
// out[Cdim][512] f16.   z<8: batch z of x (Cdim=1024, xT[s][c]);
// z=8..11: Wq/Wk/Wv (Wt[n][c]) and Wo (WoT[c][n]), Cdim=512.
// ---------------------------------------------------------------------------
__global__ __launch_bounds__(128) void transpose_cvt_kernel(
    const float* __restrict__ x,
    const float* __restrict__ Wq, const float* __restrict__ Wk,
    const float* __restrict__ Wv, const float* __restrict__ Wo,
    _Float16* __restrict__ xTh,
    _Float16* __restrict__ WqT, _Float16* __restrict__ WkT,
    _Float16* __restrict__ WvT, _Float16* __restrict__ WoT)
{
    __shared__ alignas(16) _Float16 T[64][72];

    const int tx = blockIdx.x, ty = blockIdx.y, z = blockIdx.z;
    const float* src;
    _Float16* dst;
    int Cdim;
    if (z < 8) {
        src = x + (size_t)z * C_IN * SEQ;
        dst = xTh + (size_t)z * SEQ * C_IN;
        Cdim = SEQ;
    } else {
        if (tx >= 8) return;                  // weights need only 8 col-tiles
        const float* ws_[4] = {Wq, Wk, Wv, Wo};
        _Float16*    wd_[4] = {WqT, WkT, WvT, WoT};
        src = ws_[z - 8];
        dst = wd_[z - 8];
        Cdim = C_IN;
    }
    const int tid = threadIdx.x;
    const int r0 = ty * 64, c0 = tx * 64;

    float4 t[8];
#pragma unroll
    for (int i = 0; i < 8; ++i) {
        int idx = tid + i * 128;              // 0..1023 float4 index
        int r = idx >> 4, cv = (idx & 15) * 4;
        t[i] = *(const float4*)&src[(size_t)(r0 + r) * Cdim + c0 + cv];
    }
#pragma unroll
    for (int i = 0; i < 8; ++i) {
        int idx = tid + i * 128;
        int r = idx >> 4, cv = (idx & 15) * 4;
        T[cv + 0][r] = (_Float16)t[i].x;
        T[cv + 1][r] = (_Float16)t[i].y;
        T[cv + 2][r] = (_Float16)t[i].z;
        T[cv + 3][r] = (_Float16)t[i].w;
    }
    __syncthreads();
#pragma unroll
    for (int i = 0; i < 4; ++i) {
        int idx = tid + i * 128;              // 0..511
        int row = idx >> 3, ck = idx & 7;
        *(v8h*)&dst[(size_t)(c0 + row) * C_IN + r0 + ck * 8] =
            *(const v8h*)&T[row][ck * 8];
    }
}

// ---------------------------------------------------------------------------
// Kernel 1: QKV projection.  out[s,n] = sum_c xT[s,c]*Wt[n,c] + bias[n]
// Double-buffered async LDS staging overlaps copies with WMMA.
// Q,K -> [b,h,s,d] f16.  V -> transposed [b,h,d,s] f16.
// grid: (16 s-tiles, 24 = mat*8+h, B).  block: 128 (4 waves), 64x64 tile.
// ---------------------------------------------------------------------------
__global__ __launch_bounds__(128) void qkv_kernel(
    const _Float16* __restrict__ xTh,
    const _Float16* __restrict__ WqT, const _Float16* __restrict__ WkT,
    const _Float16* __restrict__ WvT,
    const float* __restrict__ bq, const float* __restrict__ bk,
    const float* __restrict__ bv,
    _Float16* __restrict__ Qh, _Float16* __restrict__ Kh,
    _Float16* __restrict__ Vth)
{
    __shared__ alignas(16) _Float16 At[2][64][40];   // [s][k]
    __shared__ alignas(16) _Float16 Bt[2][64][40];   // [n][k]
    __shared__ alignas(16) _Float16 Ot[64][72];      // epilogue staging

    const int st  = blockIdx.x;
    const int ny  = blockIdx.y;
    const int b   = blockIdx.z;
    const int mat = ny >> 3;
    const int h   = ny & 7;
    const _Float16* Wt   = (mat == 0) ? WqT : (mat == 1) ? WkT : WvT;
    const float*    bias = (mat == 0) ? bq  : (mat == 1) ? bk  : bv;

    const int tid  = threadIdx.x;
    const int lane = tid & 31;
    const int w    = tid >> 5;
    const int s0   = st * 64;
    const _Float16* xTb = xTh + (size_t)b * SEQ * C_IN;
    const _Float16* Wh  = Wt + (size_t)h * 64 * C_IN;   // 64 rows of this head
    const size_t bh = (size_t)b * NH + h;

    // 4 async instructions per wave per stage
    auto stage = [&](int kk, int buf) {
        const int k0 = kk * 32;
#pragma unroll
        for (int i = 0; i < 2; ++i) {
            int idx = tid + i * 128;          // 0..255
            int row = idx >> 2, ck = idx & 3;
            async_b128s(&At[buf][row][ck * 8], xTb,
                        (((s0 + row) * C_IN) + k0 + ck * 8) * 2);
            async_b128s(&Bt[buf][row][ck * 8], Wh,
                        ((row * C_IN) + k0 + ck * 8) * 2);
        }
    };

    v8f acc[4];
#pragma unroll
    for (int i = 0; i < 4; ++i) acc[i] = (v8f){0.f,0.f,0.f,0.f,0.f,0.f,0.f,0.f};

    stage(0, 0);                              // prologue
    for (int kk = 0; kk < C_IN / 32; ++kk) {
        const int cur = kk & 1;
        __syncthreads();                      // buf (1-cur) free of readers
        if (kk + 1 < C_IN / 32) {
            stage(kk + 1, 1 - cur);           // overlap with this tile's WMMAs
            wait_async4();                    // oldest 4 (= buf cur) landed
        } else {
            wait_async0();
        }
        __syncthreads();                      // all waves see buf cur

        v16h a = ldfrag<40>(&At[cur][w * 16][0], lane, 0);
#pragma unroll
        for (int nf = 0; nf < 4; ++nf) {
            v16h bf = ldfrag<40>(&Bt[cur][nf * 16][0], lane, 0);
            acc[nf] = wmma_f16(a, bf, acc[nf]);
        }
    }

    // epilogue: bias, stage D fragments in LDS, coalesced b128 stores.
    const int col = lane & 15, hi = lane >> 4;
    if (mat == 2) {
        // V transposed: Ot[d][s_local]
#pragma unroll
        for (int nf = 0; nf < 4; ++nf) {
            const int d = nf * 16 + col;
            const float bb = bias[h * 64 + d];
#pragma unroll
            for (int r = 0; r < 8; ++r)
                Ot[d][w * 16 + r + hi * 8] = (_Float16)(acc[nf][r] + bb);
        }
        __syncthreads();
#pragma unroll
        for (int i = 0; i < 4; ++i) {
            int idx = tid + i * 128;
            int row = idx >> 3, ck = idx & 7;
            *(v8h*)(Vth + (bh * DHEAD + row) * SEQ + s0 + ck * 8) =
                *(const v8h*)&Ot[row][ck * 8];
        }
    } else {
        _Float16* dst = (mat == 0) ? Qh : Kh;
#pragma unroll
        for (int nf = 0; nf < 4; ++nf) {
            const int d = nf * 16 + col;
            const float bb = bias[h * 64 + d];
#pragma unroll
            for (int r = 0; r < 8; ++r)
                Ot[w * 16 + r + hi * 8][d] = (_Float16)(acc[nf][r] + bb);
        }
        __syncthreads();
#pragma unroll
        for (int i = 0; i < 4; ++i) {
            int idx = tid + i * 128;
            int row = idx >> 3, ck = idx & 7;
            *(v8h*)(dst + (bh * SEQ + s0 + row) * DHEAD + ck * 8) =
                *(const v8h*)&Ot[row][ck * 8];
        }
    }
}

// ---------------------------------------------------------------------------
// Kernel 2: flash attention for one (b, h, 64-query tile).
// Double-buffered async K/V staging (8 async ops in flight per wave).
// ---------------------------------------------------------------------------
__global__ __launch_bounds__(128) void attn_kernel(
    const _Float16* __restrict__ Qh, const _Float16* __restrict__ Kh,
    const _Float16* __restrict__ Vth, _Float16* __restrict__ AO)
{
    __shared__ alignas(16) _Float16 Qs[64][72];         // [q][d]; out staging too
    __shared__ alignas(16) _Float16 Ks[2][64][72];      // [k][d]
    __shared__ alignas(16) _Float16 Vs[2][64][72];      // [d][k]  (V transposed)
    __shared__ alignas(16) _Float16 Ps[4][16][72];      // per-wave P strip [q][k]

    const int qt = blockIdx.x, h = blockIdx.y, b = blockIdx.z;
    const int tid = threadIdx.x, lane = tid & 31, w = tid >> 5;
    const size_t bh = (size_t)b * NH + h;
    const _Float16* Qg = Qh  + bh * SEQ * DHEAD;
    const _Float16* Kg = Kh  + bh * SEQ * DHEAD;
    const _Float16* Vg = Vth + bh * DHEAD * SEQ;

    // 8 async instructions per wave per stage
    auto stage_kv = [&](int kb, int buf) {
#pragma unroll
        for (int i = 0; i < 4; ++i) {
            int idx = tid + i * 128;
            int row = idx >> 3, ck = idx & 7;
            async_b128s(&Ks[buf][row][ck * 8], Kg,
                        (((kb * 64 + row) * DHEAD) + ck * 8) * 2);
            async_b128s(&Vs[buf][row][ck * 8], Vg,
                        ((row * SEQ) + kb * 64 + ck * 8) * 2);
        }
    };

    // async-load Q tile + prologue K/V tile
#pragma unroll
    for (int i = 0; i < 4; ++i) {
        int idx = tid + i * 128;
        int row = idx >> 3, ck = idx & 7;
        async_b128s(&Qs[row][ck * 8], Qg,
                    (((qt * 64 + row) * DHEAD) + ck * 8) * 2);
    }
    stage_kv(0, 0);
    wait_async0();
    __syncthreads();
    v16h aQ0 = ldfrag<72>(&Qs[w * 16][0], lane, 0);
    v16h aQ1 = ldfrag<72>(&Qs[w * 16][0], lane, 32);

    float mrow[8], lrow[8];
    v8f o[4];
#pragma unroll
    for (int r = 0; r < 8; ++r) { mrow[r] = -1e30f; lrow[r] = 0.f; }
#pragma unroll
    for (int i = 0; i < 4; ++i) o[i] = (v8f){0.f,0.f,0.f,0.f,0.f,0.f,0.f,0.f};

    const int col = lane & 15, hi = lane >> 4;

    for (int kb = 0; kb < SEQ / 64; ++kb) {
        const int cur = kb & 1;
        __syncthreads();                  // buf (1-cur) free of readers
        if (kb + 1 < SEQ / 64) {
            stage_kv(kb + 1, 1 - cur);    // overlap with this tile's compute
            wait_async8();                // oldest 8 (= buf cur) landed
        } else {
            wait_async0();
        }
        __syncthreads();                  // all waves see buf cur

        // scores: 16x64 strip = Q(16x64) * K^T, 4 N-frags x 2 K-steps
        v8f sc[4];
#pragma unroll
        for (int nf = 0; nf < 4; ++nf) {
            v8f a = (v8f){0.f,0.f,0.f,0.f,0.f,0.f,0.f,0.f};
            a = wmma_f16(aQ0, ldfrag<72>(&Ks[cur][nf * 16][0], lane, 0),  a);
            a = wmma_f16(aQ1, ldfrag<72>(&Ks[cur][nf * 16][0], lane, 32), a);
            sc[nf] = a;
        }
#pragma unroll
        for (int nf = 0; nf < 4; ++nf)
#pragma unroll
            for (int r = 0; r < 8; ++r) sc[nf][r] *= 0.125f;   // 1/sqrt(64)

        // online softmax (rows live in one 16-lane half -> xor masks 1..8)
        float mnew[8], alpha[8];
#pragma unroll
        for (int r = 0; r < 8; ++r) {
            float mv = fmaxf(fmaxf(sc[0][r], sc[1][r]), fmaxf(sc[2][r], sc[3][r]));
#pragma unroll
            for (int m = 1; m <= 8; m <<= 1) mv = fmaxf(mv, __shfl_xor(mv, m));
            mnew[r]  = fmaxf(mrow[r], mv);
            alpha[r] = __expf(mrow[r] - mnew[r]);
            mrow[r]  = mnew[r];
        }
        float psum[8];
#pragma unroll
        for (int r = 0; r < 8; ++r) psum[r] = 0.f;
#pragma unroll
        for (int nf = 0; nf < 4; ++nf)
#pragma unroll
            for (int r = 0; r < 8; ++r) {
                float pv = __expf(sc[nf][r] - mnew[r]);
                sc[nf][r] = pv;
                psum[r] += pv;
            }
#pragma unroll
        for (int r = 0; r < 8; ++r) {
            float t = psum[r];
#pragma unroll
            for (int m = 1; m <= 8; m <<= 1) t += __shfl_xor(t, m);
            lrow[r] = lrow[r] * alpha[r] + t;
        }
#pragma unroll
        for (int nf = 0; nf < 4; ++nf)
#pragma unroll
            for (int r = 0; r < 8; ++r) o[nf][r] *= alpha[r];

        // P: C/D layout -> per-wave LDS -> reload in A layout
        _Float16* pb = &Ps[w][0][0];
#pragma unroll
        for (int nf = 0; nf < 4; ++nf)
#pragma unroll
            for (int r = 0; r < 8; ++r)
                pb[(r + hi * 8) * 72 + nf * 16 + col] = (_Float16)sc[nf][r];
        asm volatile("s_wait_dscnt 0" ::: "memory");
        v16h aP0 = ldfrag<72>(pb, lane, 0);
        v16h aP1 = ldfrag<72>(pb, lane, 32);

        // O += P(16x64) * V(64x64), B-frags from transposed V tile
#pragma unroll
        for (int nf = 0; nf < 4; ++nf) {
            o[nf] = wmma_f16(aP0, ldfrag<72>(&Vs[cur][nf * 16][0], lane, 0),  o[nf]);
            o[nf] = wmma_f16(aP1, ldfrag<72>(&Vs[cur][nf * 16][0], lane, 32), o[nf]);
        }
    }

    // normalize, stage through LDS (reuse Qs), coalesced b128 stores
#pragma unroll
    for (int nf = 0; nf < 4; ++nf)
#pragma unroll
        for (int r = 0; r < 8; ++r) {
            const float inv = 1.0f / lrow[r];
            Qs[w * 16 + r + hi * 8][nf * 16 + col] = (_Float16)(o[nf][r] * inv);
        }
    __syncthreads();
#pragma unroll
    for (int i = 0; i < 4; ++i) {
        int idx = tid + i * 128;
        int row = idx >> 3, ck = idx & 7;
        *(v8h*)(AO + ((size_t)b * SEQ + qt * 64 + row) * C_IN + h * 64 + ck * 8) =
            *(const v8h*)&Qs[row][ck * 8];
    }
}

// ---------------------------------------------------------------------------
// Kernel 3: out[b][c][s] = attnOut[b][s][:] . Wo[:][c] + bo[c] + x[b][c][s]
// Double-buffered async staging; v8f vector epilogue.
// ---------------------------------------------------------------------------
__global__ __launch_bounds__(128) void proj_kernel(
    const _Float16* __restrict__ AO, const _Float16* __restrict__ WoT,
    const float* __restrict__ bo, const float* __restrict__ x,
    float* __restrict__ out)
{
    __shared__ alignas(16) _Float16 At[2][64][40];   // [s][k]
    __shared__ alignas(16) _Float16 Bt[2][64][40];   // [c][k]

    const int st = blockIdx.x, ct = blockIdx.y, b = blockIdx.z;
    const int tid = threadIdx.x, lane = tid & 31, w = tid >> 5;
    const int s0 = st * 64, c0 = ct * 64;
    const _Float16* AOb = AO + (size_t)b * SEQ * C_IN;
    const _Float16* Woc = WoT + (size_t)c0 * C_IN;

    auto stage = [&](int kk, int buf) {
        const int k0 = kk * 32;
#pragma unroll
        for (int i = 0; i < 2; ++i) {
            int idx = tid + i * 128;                // 0..255
            int row = idx >> 2, ck = idx & 3;
            async_b128s(&At[buf][row][ck * 8], AOb,
                        (((s0 + row) * C_IN) + k0 + ck * 8) * 2);
            async_b128s(&Bt[buf][row][ck * 8], Woc,
                        ((row * C_IN) + k0 + ck * 8) * 2);
        }
    };

    v8f acc[4];
#pragma unroll
    for (int i = 0; i < 4; ++i) acc[i] = (v8f){0.f,0.f,0.f,0.f,0.f,0.f,0.f,0.f};

    stage(0, 0);
    for (int kk = 0; kk < C_IN / 32; ++kk) {
        const int cur = kk & 1;
        __syncthreads();
        if (kk + 1 < C_IN / 32) {
            stage(kk + 1, 1 - cur);
            wait_async4();
        } else {
            wait_async0();
        }
        __syncthreads();

        v16h a = ldfrag<40>(&At[cur][w * 16][0], lane, 0);
#pragma unroll
        for (int nf = 0; nf < 4; ++nf) {
            v16h bf = ldfrag<40>(&Bt[cur][nf * 16][0], lane, 0);
            acc[nf] = wmma_f16(a, bf, acc[nf]);
        }
    }

    // epilogue: rows s are contiguous -> vectorized residual + bias + store
    const int col = lane & 15, hi = lane >> 4;
#pragma unroll
    for (int nf = 0; nf < 4; ++nf) {
        const int c = c0 + nf * 16 + col;
        const float bb = bo[c];
        const size_t base = ((size_t)b * C_IN + c) * SEQ + s0 + w * 16 + hi * 8;
        v8f xv = *(const v8f*)&x[base];
        v8f res = acc[nf] + bb + xv;
        *(v8f*)&out[base] = res;
    }
}

// ---------------------------------------------------------------------------
extern "C" void kernel_launch(void* const* d_in, const int* in_sizes, int n_in,
                              void* d_out, int out_size, void* d_ws, size_t ws_size,
                              hipStream_t stream) {
    const float* x  = (const float*)d_in[0];
    const float* Wq = (const float*)d_in[1];
    const float* bq = (const float*)d_in[2];
    const float* Wk = (const float*)d_in[3];
    const float* bk = (const float*)d_in[4];
    const float* Wv = (const float*)d_in[5];
    const float* bv = (const float*)d_in[6];
    const float* Wo = (const float*)d_in[7];
    const float* bo = (const float*)d_in[8];
    float* out = (float*)d_out;

    char* ws = (char*)d_ws;
    _Float16* Qh  = (_Float16*)(ws);                   // [B,NH,S,D]  8 MB
    _Float16* Kh  = (_Float16*)(ws + (8ull  << 20));   // [B,NH,S,D]  8 MB
    _Float16* Vth = (_Float16*)(ws + (16ull << 20));   // [B,NH,D,S]  8 MB
    _Float16* AO  = (_Float16*)(ws + (24ull << 20));   // [B,S,C]     8 MB
    _Float16* xTh = (_Float16*)(ws + (32ull << 20));   // [B,S,C]     8 MB
    _Float16* WqT = (_Float16*)(ws + (40ull << 20));   // [N,C] 512 KB
    _Float16* WkT = (_Float16*)(ws + (40ull << 20) + (512ull << 10));
    _Float16* WvT = (_Float16*)(ws + (41ull << 20));
    _Float16* WoT = (_Float16*)(ws + (41ull << 20) + (512ull << 10));

    transpose_cvt_kernel<<<dim3(16, 8, 12), 128, 0, stream>>>(
        x, Wq, Wk, Wv, Wo, xTh, WqT, WkT, WvT, WoT);
    qkv_kernel<<<dim3(16, 24, 8), 128, 0, stream>>>(
        xTh, WqT, WkT, WvT, bq, bk, bv, Qh, Kh, Vth);
    attn_kernel<<<dim3(16, 8, 8), 128, 0, stream>>>(Qh, Kh, Vth, AO);
    proj_kernel<<<dim3(16, 8, 8), 128, 0, stream>>>(AO, WoT, bo, x, out);
}
